// SSR_19275813225061
// MI455X (gfx1250) — matrix operand order repaired and checked by reading, not compiled, not run on hardware
//
#include <hip/hip_runtime.h>
#include <hip/hip_bf16.h>

// ---------------------------------------------------------------------------
// Problem constants (from reference)
// ---------------------------------------------------------------------------
#define FD     2048          // FEAT_DIM
#define KK     1024          // K
#define NROW   1025          // K + 1 rows of feat/sim
#define DIN    2049          // 2K+1
#define HD     2048          // hidden
#define KP_DIN 2080          // DIN padded to multiple of 32 (zero-filled)
#define KP_NR  1056          // NROW padded to multiple of 32 (zero-filled)
#define LR     0.001f
#define EPSIN  1e-5f

typedef __attribute__((ext_vector_type(16))) _Float16 v16h;
typedef __attribute__((ext_vector_type(8)))  float    v8f;

union FragA { v16h h; unsigned int u[8]; };

// CDNA5 async global->LDS DMA (ASYNCcnt-tracked); VDST = per-lane LDS byte
// offset, VADDR = per-lane 64-bit global address.
__device__ __forceinline__ void async_load_b128(unsigned lds_off, const void* gaddr)
{
    asm volatile("global_load_async_to_lds_b128 %0, %1, off"
                 :: "v"(lds_off), "v"(gaddr) : "memory");
}
__device__ __forceinline__ void wait_async0()
{
    asm volatile("s_wait_asynccnt 0x0" ::: "memory");
}

// ---------------------------------------------------------------------------
// WMMA GEMM:  C[M,N](f32) = A[M,K](f16) * op(B) + bias, optional ReLU
//   BT=true : B is (N,K) row-major  -> C = A * B^T
//   BT=false: B is (K,N) row-major  -> C = A * B      (requires N % 8 == 0)
// Host guarantees: lda/ldb multiple of 8 halves; Kd multiple of 32 with
// zero-filled K-pad in both operands -> no K bounds checks. Row indices are
// clamped (garbage rows only feed discarded outputs).
// 256 threads = 8 waves; block tile 128x64; wave tile 32x32 (4 WMMA/K-tile).
// Double-buffered LDS; A (and NT B) staged via global_load_async_to_lds_b128.
// ---------------------------------------------------------------------------
#define BM 128
#define BN 64
#define KT 32
#define LDS_K 40             // LDS row stride in halves (80B: 16B-aligned)

template<bool BT>
__global__ __launch_bounds__(256)
void gemm_wmma(const _Float16* __restrict__ A, int lda,
               const _Float16* __restrict__ B, int ldb,
               float* __restrict__ C, int ldc,
               int M, int N, int Kd,
               const float* __restrict__ bias, int relu)
{
    __shared__ _Float16 As[2][BM][LDS_K];
    __shared__ _Float16 Bs[2][BN][LDS_K];

    const int tid  = threadIdx.x;
    const int lane = tid & 31;            // wave32
    const int wid  = tid >> 5;
    const int wm   = wid & 3;             // 0..3 -> 32-row wave tile
    const int wn   = wid >> 2;            // 0..1 -> 32-col wave tile
    const int m0   = blockIdx.y * BM;
    const int n0   = blockIdx.x * BN;

    v8f acc00 = {}, acc01 = {}, acc10 = {}, acc11 = {};

    // staging maps (one 16B move = 8 halves per op, fully aligned)
    const int arow = tid >> 2;            // 0..63
    const int ak   = (tid & 3) * 8;       // 0,8,16,24
    const int bk   = tid >> 3;            // 0..31 (NN: k)
    const int bn8  = (tid & 7) * 8;       // 0..56 (NN: n chunk)

    int gmA0 = m0 + arow;       if (gmA0 > M - 1) gmA0 = M - 1;
    int gmA1 = m0 + arow + 64;  if (gmA1 > M - 1) gmA1 = M - 1;
    int gnB  = n0 + arow;       if (gnB  > N - 1) gnB  = N - 1;

    const int nkt = Kd / KT;              // Kd is a multiple of KT

    auto stage = [&](int t, int nb) {
        const int k0 = t * KT;
        async_load_b128((unsigned)(size_t)&As[nb][arow][ak],
                        &A[(size_t)gmA0 * lda + k0 + ak]);
        async_load_b128((unsigned)(size_t)&As[nb][arow + 64][ak],
                        &A[(size_t)gmA1 * lda + k0 + ak]);
        if (BT) {
            async_load_b128((unsigned)(size_t)&Bs[nb][arow][ak],
                            &B[(size_t)gnB * ldb + k0 + ak]);
        } else {
            // B[k][n]: vector read along n, scatter into Bs[n][k]
            uint4 v = *(const uint4*)&B[(size_t)(k0 + bk) * ldb + n0 + bn8];
            const _Float16* h = (const _Float16*)&v;
            #pragma unroll
            for (int j = 0; j < 8; ++j) Bs[nb][bn8 + j][bk] = h[j];
        }
        if (t + 1 < nkt) {    // L2 warm-up one more tile ahead
            __builtin_prefetch(&A[(size_t)gmA0 * lda + k0 + KT + ak], 0, 1);
            __builtin_prefetch(&A[(size_t)gmA1 * lda + k0 + KT + ak], 0, 1);
        }
    };

    stage(0, 0);
    for (int t = 0; t < nkt; ++t) {
        const int buf = t & 1;
        wait_async0();        // this wave's DMA into `buf` (and its NN scatter) done
        __syncthreads();      // everyone's staging of `buf` visible; prior reads of buf^1 drained
        if (t + 1 < nkt) stage(t + 1, buf ^ 1);

        // ---- fragments (ISA 16-bit A 16x32 / B 32x16 lane layouts)
        FragA af0, af1, bf0, bf1;
        const int hi = lane >> 4;
        const int am = lane & 15;
        #pragma unroll
        for (int v = 0; v < 8; ++v) {
            const int k = (v < 4) ? (hi * 8 + v * 2) : (16 + hi * 8 + (v - 4) * 2);
            af0.u[v] = *(const unsigned int*)&As[buf][wm * 32 + am][k];
            af1.u[v] = *(const unsigned int*)&As[buf][wm * 32 + 16 + am][k];
        }
        #pragma unroll
        for (int t2 = 0; t2 < 8; ++t2) {
            const int k = (t2 < 4) ? (hi * 8 + t2 * 2) : (16 + hi * 8 + (t2 - 4) * 2);
            bf0.u[t2] = *(const unsigned int*)&Bs[buf][wn * 32 + am][k];
            bf1.u[t2] = *(const unsigned int*)&Bs[buf][wn * 32 + 16 + am][k];
        }
        acc00 = __builtin_amdgcn_wmma_f32_16x16x32_f16(false, af0.h, false, bf0.h, (short)0, acc00, false, false);
        acc01 = __builtin_amdgcn_wmma_f32_16x16x32_f16(false, af0.h, false, bf1.h, (short)0, acc01, false, false);
        acc10 = __builtin_amdgcn_wmma_f32_16x16x32_f16(false, af1.h, false, bf0.h, (short)0, acc10, false, false);
        acc11 = __builtin_amdgcn_wmma_f32_16x16x32_f16(false, af1.h, false, bf1.h, (short)0, acc11, false, false);
    }

    // ---- epilogue: C 16x16 f32 layout: lane -> N=lane%16, VGPR r -> M=(lane/16)*8+r
    const int cn = lane & 15;
    const int mb = (lane >> 4) * 8;
    #pragma unroll
    for (int mi = 0; mi < 2; ++mi) {
        #pragma unroll
        for (int ni = 0; ni < 2; ++ni) {
            const v8f& a = (mi == 0) ? (ni == 0 ? acc00 : acc01)
                                     : (ni == 0 ? acc10 : acc11);
            const int gn = n0 + wn * 32 + ni * 16 + cn;
            if (gn >= N) continue;
            const float bv = bias ? bias[gn] : 0.f;
            #pragma unroll
            for (int r = 0; r < 8; ++r) {
                const int gm = m0 + wm * 32 + mi * 16 + mb + r;
                if (gm < M) {
                    float v = a[r] + bv;
                    if (relu) v = fmaxf(v, 0.f);
                    C[(size_t)gm * ldc + gn] = v;
                }
            }
        }
    }
}

// ---------------------------------------------------------------------------
// Row normalize: U = f/max(||f||,1e-12); writes f32 U, f16 shadow, norms.
// ---------------------------------------------------------------------------
__global__ __launch_bounds__(256)
void row_normalize(const float* __restrict__ fq, const float* __restrict__ fd,
                   float* __restrict__ U, _Float16* __restrict__ Uh,
                   float* __restrict__ norms)
{
    const int r = blockIdx.x;
    const float* src = (r == 0) ? fq : fd + (size_t)(r - 1) * FD;
    __shared__ float red[256];
    float s = 0.f;
    for (int i = threadIdx.x; i < FD; i += 256) { float v = src[i]; s += v * v; }
    red[threadIdx.x] = s; __syncthreads();
    for (int st = 128; st > 0; st >>= 1) {
        if (threadIdx.x < st) red[threadIdx.x] += red[threadIdx.x + st];
        __syncthreads();
    }
    const float nm  = fmaxf(sqrtf(red[0]), 1e-12f);
    const float inv = 1.f / nm;
    for (int i = threadIdx.x; i < FD; i += 256) {
        const float v = src[i] * inv;
        U [(size_t)r * FD + i] = v;
        Uh[(size_t)r * FD + i] = (_Float16)v;
    }
    if (threadIdx.x == 0) norms[r] = nm;
}

// ---------------------------------------------------------------------------
// Per-row full descending argsort (top_k with K == ncols) via bitonic sort in
// LDS, fused with building inp (1025 x 2049). Tie-break: lower index first.
// ---------------------------------------------------------------------------
__global__ __launch_bounds__(256)
void sort_build_inp(const float* __restrict__ sim,
                    float* __restrict__ inp, int* __restrict__ idxb)
{
    __shared__ float v[KK];
    __shared__ int   id[KK];
    const int i = blockIdx.x;
    for (int m = threadIdx.x; m < KK; m += 256) {
        v[m]  = sim[(size_t)i * NROW + m + 1];   // sim_d[i,m] = sim[i, m+1]
        id[m] = m;
    }
    __syncthreads();
    for (int size = 2; size <= KK; size <<= 1) {
        for (int stride = size >> 1; stride > 0; stride >>= 1) {
            for (int t = threadIdx.x; t < KK / 2; t += 256) {
                const int lo  = t & (stride - 1);
                const int pos = ((t ^ lo) << 1) | lo;     // bitonic pair base
                const int par = pos + stride;
                const bool desc = ((pos & size) == 0);
                const float va = v[pos], vb = v[par];
                const int   ia = id[pos], ib = id[par];
                const bool b_first = (vb > va) || (vb == va && ib < ia);
                const bool a_first = (va > vb) || (va == vb && ia < ib);
                if (desc ? b_first : a_first) {
                    v[pos] = vb; v[par] = va; id[pos] = ib; id[par] = ia;
                }
            }
            __syncthreads();
        }
    }
    for (int m = threadIdx.x; m < KK; m += 256) {
        const int c = id[m];
        inp [(size_t)i * DIN + 1 + m]      = v[m];
        inp [(size_t)i * DIN + 1 + KK + m] = sim[c + 1];   // sim[0, c+1]
        idxb[(size_t)i * KK + m]           = c;
    }
    if (threadIdx.x == 0) inp[(size_t)i * DIN] = sim[i];    // sim[0, i]
}

// ---------------------------------------------------------------------------
// Instance norm over last dim (scalar affine) -> f16 with zero-filled K-pad.
// ---------------------------------------------------------------------------
__global__ __launch_bounds__(256)
void instnorm_f16(const float* __restrict__ x, int L, int outld,
                  const float* __restrict__ w, const float* __restrict__ b,
                  _Float16* __restrict__ out)
{
    const int row = blockIdx.x;
    const float* xr = x + (size_t)row * L;
    __shared__ float s1[256], s2[256];
    float a = 0.f, q = 0.f;
    for (int i = threadIdx.x; i < L; i += 256) { float v = xr[i]; a += v; q += v * v; }
    s1[threadIdx.x] = a; s2[threadIdx.x] = q; __syncthreads();
    for (int st = 128; st > 0; st >>= 1) {
        if (threadIdx.x < st) { s1[threadIdx.x] += s1[threadIdx.x + st];
                                s2[threadIdx.x] += s2[threadIdx.x + st]; }
        __syncthreads();
    }
    const float mean = s1[0] / (float)L;
    float var = s2[0] / (float)L - mean * mean;
    if (var < 0.f) var = 0.f;
    const float rstd = rsqrtf(var + EPSIN);
    const float ww = w[0], bb = b[0];
    for (int i = threadIdx.x; i < outld; i += 256) {
        const float y = (i < L) ? ((xr[i] - mean) * rstd) * ww + bb : 0.f;
        out[(size_t)row * outld + i] = (_Float16)y;
    }
}

// ---------------------------------------------------------------------------
// Gather-VJP scatter:  g (1025x2049) -> G (1025x1025)
// ---------------------------------------------------------------------------
__global__ void zero_f32(float* __restrict__ p, int n)
{
    const int t = blockIdx.x * 256 + threadIdx.x;
    if (t < n) p[t] = 0.f;
}

__global__ void zero_f16(_Float16* __restrict__ p, int n)
{
    const int t = blockIdx.x * 256 + threadIdx.x;
    if (t < n) p[t] = (_Float16)0.f;
}

__global__ void scatter_g(const float* __restrict__ g,
                          const int* __restrict__ idxb, float* __restrict__ G)
{
    const int t = blockIdx.x * 256 + threadIdx.x;
    if (t >= NROW * KK) return;
    const int i = t >> 10;
    const int m = t & (KK - 1);
    const int c = idxb[t];
    atomicAdd(&G[(size_t)i * NROW + c + 1], g[(size_t)i * DIN + 1 + m]);       // d sim[i,c+1]
    atomicAdd(&G[c + 1],                    g[(size_t)i * DIN + 1 + KK + m]);  // d sim[0,c+1]
    if (m == 0) atomicAdd(&G[i],            g[(size_t)i * DIN]);               // d sim[0,i]
}

// S = G + G^T in f16, leading dim KP_NR with zero-filled pad columns.
__global__ void build_s_f16(const float* __restrict__ G, _Float16* __restrict__ Sh)
{
    const int t = blockIdx.x * 256 + threadIdx.x;
    if (t >= NROW * KP_NR) return;
    const int i = t / KP_NR, j = t - i * KP_NR;
    float s = 0.f;
    if (j < NROW) s = G[(size_t)i * NROW + j] + G[(size_t)j * NROW + i];
    Sh[t] = (_Float16)s;
}

// ---------------------------------------------------------------------------
// Normalize backward + SGD update:  f -= LR * (dU - u*(u.dU)) / n
// ---------------------------------------------------------------------------
__global__ __launch_bounds__(256)
void norm_bwd_update(const float* __restrict__ U, const float* __restrict__ dU,
                     const float* __restrict__ norms, float* __restrict__ target,
                     int urow0)
{
    const int r = urow0 + blockIdx.x;
    const float* u  = U  + (size_t)r * FD;
    const float* du = dU + (size_t)r * FD;
    float* f = target + (size_t)blockIdx.x * FD;
    __shared__ float red[256];
    float d = 0.f;
    for (int i = threadIdx.x; i < FD; i += 256) d += u[i] * du[i];
    red[threadIdx.x] = d; __syncthreads();
    for (int st = 128; st > 0; st >>= 1) {
        if (threadIdx.x < st) red[threadIdx.x] += red[threadIdx.x + st];
        __syncthreads();
    }
    const float dot = red[0];
    const float inv = 1.f / norms[r];
    for (int i = threadIdx.x; i < FD; i += 256)
        f[i] -= LR * ((du[i] - u[i] * dot) * inv);
}

// ---------------------------------------------------------------------------
// Final sqd[c] = u0 . u_{c+1}
// ---------------------------------------------------------------------------
__global__ __launch_bounds__(256)
void final_sqd(const float* __restrict__ U, float* __restrict__ out)
{
    const int c = blockIdx.x;
    const float* a = U;
    const float* b = U + (size_t)(c + 1) * FD;
    __shared__ float red[256];
    float d = 0.f;
    for (int i = threadIdx.x; i < FD; i += 256) d += a[i] * b[i];
    red[threadIdx.x] = d; __syncthreads();
    for (int st = 128; st > 0; st >>= 1) {
        if (threadIdx.x < st) red[threadIdx.x] += red[threadIdx.x + st];
        __syncthreads();
    }
    if (threadIdx.x == 0) out[c] = red[0];
}

// f32 (rows x cols) -> f16 (rows x dstld), pad columns zero-filled.
__global__ void cvt_pad_f16(const float* __restrict__ x, int cols,
                            _Float16* __restrict__ y, int dstld, int rows)
{
    const size_t t = (size_t)blockIdx.x * 256 + threadIdx.x;
    if (t >= (size_t)rows * dstld) return;
    const int row = (int)(t / dstld);
    const int col = (int)(t - (size_t)row * dstld);
    y[t] = (col < cols) ? (_Float16)x[(size_t)row * cols + col] : (_Float16)0.f;
}

// ---------------------------------------------------------------------------
// Host orchestration
// ---------------------------------------------------------------------------
struct DniP {
    const float *inaw, *inab, *W1, *b1, *inbw, *inbb, *W2, *b2, *incw, *incb, *W3, *b3;
    _Float16 *W1h, *W2h, *W3h;
};

extern "C" void kernel_launch(void* const* d_in, const int* in_sizes, int n_in,
                              void* d_out, int out_size, void* d_ws, size_t ws_size,
                              hipStream_t stream)
{
    (void)n_in; (void)out_size; (void)ws_size;
    const float* fq_in = (const float*)d_in[0];
    const float* fd_in = (const float*)d_in[1];

    // Nested-dict flattening order is ambiguous; detect at runtime from sizes.
    auto mapP = [&](int base, DniP& p) {
        const float* const* q = (const float* const*)(d_in + base);
        if (in_sizes[base] == 1) {   // insertion order: in_a_w,in_a_b,W1,b1,...
            p.inaw = q[0];  p.inab = q[1];  p.W1 = q[2];  p.b1 = q[3];
            p.inbw = q[4];  p.inbb = q[5];  p.W2 = q[6];  p.b2 = q[7];
            p.incw = q[8];  p.incb = q[9];  p.W3 = q[10]; p.b3 = q[11];
        } else {                     // jax-pytree alphabetical order
            p.W1 = q[0];  p.W2 = q[1];  p.W3 = q[2];
            p.b1 = q[3];  p.b2 = q[4];  p.b3 = q[5];
            p.inab = q[6];  p.inaw = q[7];  p.inbb = q[8];
            p.inbw = q[9];  p.incb = q[10]; p.incw = q[11];
        }
    };
    DniP Pq, Pd;
    mapP(2, Pq);
    mapP(14, Pd);

    // Workspace arena
    char* cur = (char*)d_ws;
    auto alloc = [&](size_t bytes) -> void* {
        void* p = (void*)cur;
        cur += (bytes + 255) & ~(size_t)255;
        return p;
    };
    float*    fqc   = (float*)   alloc((size_t)FD * 4);
    float*    fdc   = (float*)   alloc((size_t)KK * FD * 4);
    float*    U     = (float*)   alloc((size_t)NROW * FD * 4);
    _Float16* Uh    = (_Float16*)alloc((size_t)KP_NR * FD * 2);   // rows 1025..1055 zero
    float*    norms = (float*)   alloc((size_t)NROW * 4);
    float*    sim   = (float*)   alloc((size_t)NROW * NROW * 4);
    int*      idxb  = (int*)     alloc((size_t)NROW * KK * 4);
    float*    inp   = (float*)   alloc((size_t)NROW * DIN * 4);
    float*    hf32  = (float*)   alloc((size_t)NROW * HD * 4);
    _Float16* hf16  = (_Float16*)alloc((size_t)NROW * KP_DIN * 2);
    float*    gbuf  = (float*)   alloc((size_t)NROW * DIN * 4);
    float*    G     = (float*)   alloc((size_t)NROW * NROW * 4);
    _Float16* Sh    = (_Float16*)alloc((size_t)NROW * KP_NR * 2);
    float*    dU    = (float*)   alloc((size_t)NROW * FD * 4);
    Pq.W1h = (_Float16*)alloc((size_t)HD  * KP_DIN * 2);
    Pq.W2h = (_Float16*)alloc((size_t)HD  * HD     * 2);
    Pq.W3h = (_Float16*)alloc((size_t)DIN * HD     * 2);
    Pd.W1h = (_Float16*)alloc((size_t)HD  * KP_DIN * 2);
    Pd.W2h = (_Float16*)alloc((size_t)HD  * HD     * 2);
    Pd.W3h = (_Float16*)alloc((size_t)DIN * HD     * 2);

    // Working copies of features (inputs stay pristine).
    hipMemcpyAsync(fqc, fq_in, (size_t)FD * 4, hipMemcpyDeviceToDevice, stream);
    hipMemcpyAsync(fdc, fd_in, (size_t)KK * FD * 4, hipMemcpyDeviceToDevice, stream);

    // Zero the K-pad rows of Uh once per call (nothing else writes them).
    zero_f16<<<((KP_NR - NROW) * FD + 255) / 256, 256, 0, stream>>>(
        Uh + (size_t)NROW * FD, (KP_NR - NROW) * FD);

    // f16 weight shadows with zero-filled K pad (recomputed every call).
    auto cvt = [&](const float* s, int cols, _Float16* d, int dstld, int rows) {
        const size_t n = (size_t)rows * dstld;
        cvt_pad_f16<<<(unsigned)((n + 255) / 256), 256, 0, stream>>>(s, cols, d, dstld, rows);
    };
    cvt(Pq.W1, DIN, Pq.W1h, KP_DIN, HD);  cvt(Pq.W2, HD, Pq.W2h, HD, HD);
    cvt(Pq.W3, HD,  Pq.W3h, HD, DIN);     cvt(Pd.W1, DIN, Pd.W1h, KP_DIN, HD);
    cvt(Pd.W2, HD,  Pd.W2h, HD, HD);      cvt(Pd.W3, HD,  Pd.W3h, HD, DIN);

    const dim3 blk(256);
    const unsigned gy = (NROW + BM - 1) / BM;   // 9
    auto dni = [&](const DniP& P) {
        instnorm_f16<<<NROW, blk, 0, stream>>>(inp, DIN, KP_DIN, P.inaw, P.inab, hf16);
        gemm_wmma<true><<<dim3(HD / BN, gy), blk, 0, stream>>>(
            hf16, KP_DIN, P.W1h, KP_DIN, hf32, HD, NROW, HD, KP_DIN, P.b1, 1);
        instnorm_f16<<<NROW, blk, 0, stream>>>(hf32, HD, HD, P.inbw, P.inbb, hf16);
        gemm_wmma<true><<<dim3(HD / BN, gy), blk, 0, stream>>>(
            hf16, HD, P.W2h, HD, hf32, HD, NROW, HD, HD, P.b2, 1);
        instnorm_f16<<<NROW, blk, 0, stream>>>(hf32, HD, HD, P.incw, P.incb, hf16);
        gemm_wmma<true><<<dim3((DIN + BN - 1) / BN, gy), blk, 0, stream>>>(
            hf16, HD, P.W3h, HD, gbuf, DIN, NROW, DIN, HD, P.b3, 0);
    };

    auto half_step = [&](bool qphase, const DniP& P) {
        row_normalize<<<NROW, blk, 0, stream>>>(fqc, fdc, U, Uh, norms);
        gemm_wmma<true><<<dim3((NROW + BN - 1) / BN, gy), blk, 0, stream>>>(
            Uh, FD, Uh, FD, sim, NROW, NROW, NROW, FD, nullptr, 0);     // sim = U U^T
        sort_build_inp<<<NROW, blk, 0, stream>>>(sim, inp, idxb);
        dni(P);                                                          // g = dni_forward(inp)
        zero_f32<<<(NROW * NROW + 255) / 256, blk, 0, stream>>>(G, NROW * NROW);
        scatter_g<<<(NROW * KK) / 256, blk, 0, stream>>>(gbuf, idxb, G);
        build_s_f16<<<(NROW * KP_NR + 255) / 256, blk, 0, stream>>>(G, Sh);
        gemm_wmma<false><<<dim3(FD / BN, gy), blk, 0, stream>>>(
            Sh, KP_NR, Uh, FD, dU, FD, NROW, FD, KP_NR, nullptr, 0);     // dU = (G+G^T) U
        if (qphase)
            norm_bwd_update<<<1, blk, 0, stream>>>(U, dU, norms, fqc, 0);
        else
            norm_bwd_update<<<KK, blk, 0, stream>>>(U, dU, norms, fdc, 1);
    };

    for (int s = 0; s < 2; ++s) {   // Q_STEPS
        half_step(true,  Pq);
        half_step(false, Pd);
    }

    // Final pairwise -> sqd
    row_normalize<<<NROW, blk, 0, stream>>>(fqc, fdc, U, Uh, norms);
    final_sqd<<<KK, blk, 0, stream>>>(U, (float*)d_out);
}
